// VDL_51281909514270
// MI455X (gfx1250) — compile-verified, matrix operand order checked by reading, not run on hardware
//
#include <hip/hip_runtime.h>
#include <math.h>

// ---------------------------------------------------------------------------
// Fused  Y = act_cyclic(X @ W^T + b)  for MI455X (gfx1250, wave32, WMMA)
//   X: [B,1024] fp32   W: [1024,1024] fp32 ([out,in])   b: [1024]   Y fp32
//
// bf16 WMMA 16x16x32 (fp32 accum). Double-buffered LDS pipeline:
//   per K-step(32): block loads X-tile 128x32 + W-tile 256x32 (12 b128/thread,
//   contiguous), converts fp32->bf16 once (add 0x8000 + v_perm_b32), stores
//   into LDS in the exact WMMA operand layout; waves ds_load fragments
//   (2 b128 each) and run 16 WMMAs (wave tile 64x64, 8 waves as 2Mx4N).
// Grid N-fastest: X reuse via 192MB L2, W (4MB) resident in L2;
// HBM traffic ~= ideal 516MB ~ 22us.
// ---------------------------------------------------------------------------

typedef __attribute__((ext_vector_type(16))) __bf16        v16bf;
typedef __attribute__((ext_vector_type(8)))  float         v8f;
typedef __attribute__((ext_vector_type(8)))  unsigned int  u32x8;

#define IN_F   1024
#define OUT_F  1024
#define KSTEP  32
#define NITER  (IN_F / KSTEP)
// LDS geometry in uint (4B) units. One "lane row" = 16 bf16 = 8 uints,
// padded to 12 so lane stride is 12 banks (2-way conflicts, 16B aligned).
#define LROW   12
#define A_SZ   (8  * 32 * LROW)     // 128x32 bf16 tile  (3072 uints)
#define B_SZ   (16 * 32 * LROW)     // 256x32 bf16 tile  (6144 uints)
#define BUF_SZ (A_SZ + B_SZ)        // 9216 uints = 36 KB per buffer

// Pack two fp32 -> packed bf16 pair (hi<<16 | lo), round-to-nearest.
__device__ __forceinline__ unsigned int pack2_bf16(float lo, float hi) {
#if __has_builtin(__builtin_amdgcn_cvt_pk_bf16_f32)
    typedef __attribute__((ext_vector_type(2))) __bf16 v2bf;
    v2bf r = __builtin_amdgcn_cvt_pk_bf16_f32(lo, hi);
    return __builtin_bit_cast(unsigned int, r);
#else
    unsigned int a = __float_as_uint(hi) + 0x8000u;  // S0 -> byte codes 4..7
    unsigned int b = __float_as_uint(lo) + 0x8000u;  // S1 -> byte codes 0..3
    return __builtin_amdgcn_perm(a, b, 0x07060302u); // [hi.hi16 , lo.hi16]
#endif
}

__device__ __forceinline__ uint4 pack8(float4 p, float4 q) {
    uint4 r;
    r.x = pack2_bf16(p.x, p.y);
    r.y = pack2_bf16(p.z, p.w);
    r.z = pack2_bf16(q.x, q.y);
    r.w = pack2_bf16(q.z, q.w);
    return r;
}

__device__ __forceinline__ float apply_act(float y, int act) {
    // act: 0=tanh, 1=sin, 2=relu, 3=sigmoid  (uniform per lane: act = lane&3)
    if (act == 0) return tanhf(y);
    if (act == 1) return __sinf(y);
    if (act == 2) return fmaxf(y, 0.0f);
    return 1.0f / (1.0f + __expf(-y));
}

__global__ __launch_bounds__(256, 2) void fused_linear_cyclic_act(
    const float* __restrict__ X,
    const float* __restrict__ W,
    const float* __restrict__ bias,
    float* __restrict__ out)
{
    __shared__ unsigned int sm[2 * BUF_SZ];     // 72 KB

    const int tid  = threadIdx.x;
    const int lane = tid & 31;
    const int wid  = tid >> 5;                  // 0..7
    const int wm   = wid >> 2;                  // 0..1 : 64-row strip
    const int wn   = wid & 3;                   // 0..3 : 64-col strip
    const int l16  = lane & 15;
    const int g    = lane >> 4;

    // ---------------- producer setup (cooperative tile load) ---------------
    // A tile: 128 rows x 32 K. thread t -> row t/2, k-base (t&1)*16 (16 floats).
    const int arow = tid >> 1;
    const int aks  = (tid & 1) << 4;
    const float* __restrict__ gA = X + (size_t)(blockIdx.y * 128 + arow) * IN_F + aks;
    // B tile: 256 rows x 32 K. thread t -> row t (32 floats).
    const float* __restrict__ gB = W + (size_t)(blockIdx.x * 256 + tid) * IN_F;

    // LDS store addresses (uint units), WMMA operand layout:
    //   laneRow(tile, g, m) = (tile*32 + g*16 + m) * LROW ; element e at +e/2.
    const int amt = arow >> 4, am = arow & 15;
    const int aeo = (aks >> 4) << 2;                            // 0 | 4 uints
    const unsigned aSt0 =        (unsigned)((amt * 32      + am) * LROW + aeo); // g=0
    const unsigned aSt1 =        (unsigned)((amt * 32 + 16 + am) * LROW + aeo); // g=1
    const int bnt = tid >> 4, bm = tid & 15;
    const unsigned bSt0 = A_SZ + (unsigned)((bnt * 32      + bm) * LROW);       // g=0
    const unsigned bSt1 = A_SZ + (unsigned)((bnt * 32 + 16 + bm) * LROW);       // g=1

    // ---------------- consumer setup (fragment addresses) ------------------
    unsigned aLd[4], bLd[4];
#pragma unroll
    for (int mt = 0; mt < 4; ++mt) aLd[mt] =        (unsigned)(((wm * 4 + mt) * 32 + lane) * LROW);
#pragma unroll
    for (int nt = 0; nt < 4; ++nt) bLd[nt] = A_SZ + (unsigned)(((wn * 4 + nt) * 32 + lane) * LROW);

    v8f acc[4][4];
#pragma unroll
    for (int mt = 0; mt < 4; ++mt)
#pragma unroll
        for (int nt = 0; nt < 4; ++nt)
            acc[mt][nt] = (v8f){0.f, 0.f, 0.f, 0.f, 0.f, 0.f, 0.f, 0.f};

    // ---------------- software pipeline: preload K-step 0 ------------------
    float4 ra[4], rb[8];
#pragma unroll
    for (int i = 0; i < 4; ++i) ra[i] = *(const float4*)(gA + 4 * i);
#pragma unroll
    for (int i = 0; i < 8; ++i) rb[i] = *(const float4*)(gB + 4 * i);

    for (int it = 0; it < NITER; ++it) {
        const unsigned base = (unsigned)(it & 1) * BUF_SZ;

        // convert once per element, stage into LDS in WMMA layout
        *(uint4*)&sm[base + aSt0]     = pack8(ra[0], ra[1]);  // k=aks+0..7
        *(uint4*)&sm[base + aSt1]     = pack8(ra[2], ra[3]);  // k=aks+8..15
        *(uint4*)&sm[base + bSt0]     = pack8(rb[0], rb[1]);  // k= 0..7
        *(uint4*)&sm[base + bSt1]     = pack8(rb[2], rb[3]);  // k= 8..15
        *(uint4*)&sm[base + bSt0 + 4] = pack8(rb[4], rb[5]);  // k=16..23
        *(uint4*)&sm[base + bSt1 + 4] = pack8(rb[6], rb[7]);  // k=24..31

        // issue next K-step's global loads (overlap barrier + DS + WMMA)
        if (it + 1 < NITER) {
            const int kk = (it + 1) * KSTEP;
#pragma unroll
            for (int i = 0; i < 4; ++i) ra[i] = *(const float4*)(gA + kk + 4 * i);
#pragma unroll
            for (int i = 0; i < 8; ++i) rb[i] = *(const float4*)(gB + kk + 4 * i);
        }

        __syncthreads();   // s_wait_dscnt 0 + barrier: stores visible,
                           // prior-iteration frag reads complete

        v16bf af[4], bf4[4];
#pragma unroll
        for (int mt = 0; mt < 4; ++mt) {
            uint4 lo = *(const uint4*)&sm[base + aLd[mt]];
            uint4 hi = *(const uint4*)&sm[base + aLd[mt] + 4];
            u32x8 u;
            u[0] = lo.x; u[1] = lo.y; u[2] = lo.z; u[3] = lo.w;
            u[4] = hi.x; u[5] = hi.y; u[6] = hi.z; u[7] = hi.w;
            af[mt] = __builtin_bit_cast(v16bf, u);
        }
#pragma unroll
        for (int nt = 0; nt < 4; ++nt) {
            uint4 lo = *(const uint4*)&sm[base + bLd[nt]];
            uint4 hi = *(const uint4*)&sm[base + bLd[nt] + 4];
            u32x8 u;
            u[0] = lo.x; u[1] = lo.y; u[2] = lo.z; u[3] = lo.w;
            u[4] = hi.x; u[5] = hi.y; u[6] = hi.z; u[7] = hi.w;
            bf4[nt] = __builtin_bit_cast(v16bf, u);
        }

#pragma unroll
        for (int mt = 0; mt < 4; ++mt)
#pragma unroll
            for (int nt = 0; nt < 4; ++nt)
                acc[mt][nt] = __builtin_amdgcn_wmma_f32_16x16x32_bf16(
                    false, af[mt], false, bf4[nt],
                    (short)0, acc[mt][nt], false, false);
    }

    // ---------------- epilogue: bias + cyclic activation + store -----------
    // C/D layout: VGPR r -> row (r + 8*(lane/16)), column (lane&15).
    const int row0 = blockIdx.y * 128 + wm * 64;
    const int col0 = blockIdx.x * 256 + wn * 64;
    const int act  = l16 & 3;
#pragma unroll
    for (int nt = 0; nt < 4; ++nt) {
        const int col = col0 + nt * 16 + l16;
        const float bv = bias[col];
#pragma unroll
        for (int mt = 0; mt < 4; ++mt) {
            const int rbase = row0 + mt * 16 + 8 * g;
#pragma unroll
            for (int r = 0; r < 8; ++r) {
                float y = acc[mt][nt][r] + bv;
                y = apply_act(y, act);
                out[(size_t)(rbase + r) * OUT_F + col] = y;
            }
        }
    }
}

extern "C" void kernel_launch(void* const* d_in, const int* in_sizes, int n_in,
                              void* d_out, int out_size, void* d_ws, size_t ws_size,
                              hipStream_t stream) {
    const float* X    = (const float*)d_in[0];
    const float* W    = (const float*)d_in[1];
    const float* bias = (const float*)d_in[2];
    float* out        = (float*)d_out;

    const int Brows = in_sizes[0] / IN_F;   // 65536

    dim3 grid(OUT_F / 256, Brows / 128, 1); // N-fastest for L2 reuse of X
    dim3 block(256, 1, 1);
    fused_linear_cyclic_act<<<grid, block, 0, stream>>>(X, W, bias, out);
}
